// NADE_72722386256342
// MI455X (gfx1250) — matrix operand order repaired and checked by reading, not compiled, not run on hardware
//
#include <hip/hip_runtime.h>
#include <math.h>

typedef __attribute__((ext_vector_type(16))) _Float16 v16h;
typedef __attribute__((ext_vector_type(8)))  _Float16 v8h;
typedef __attribute__((ext_vector_type(2)))  _Float16 v2h;
typedef __attribute__((ext_vector_type(2)))  __fp16   v2hf;   // cvt_pkrtz result type
typedef __attribute__((ext_vector_type(8)))  float    v8f;
typedef __attribute__((ext_vector_type(4)))  float    v4f;
typedef __attribute__((ext_vector_type(2)))  float    v2f;

#define NADE_D   784
#define NADE_H   500
#define NADE_HP  512
#define NADE_B   8192
#define TILE_ROWS 16
#define NWAVES    4      // waves per 16-row batch tile (H split 4 ways)
#define NCHUNK    4      // K=32 chunks per wave -> 128 H positions per wave
#define THREADS   (NWAVES * 32)
#define PFDIST    4      // prefetch distance in d-steps

// Split workgroup barrier with a DS-only fence: unlike __syncthreads(), this
// does NOT drain LOADcnt, so pipelined global loads stay outstanding across it.
__device__ __forceinline__ void wg_barrier_ds_only() {
  asm volatile("s_wait_dscnt 0x0\n\t"
               "s_barrier_signal -1\n\t"
               "s_barrier_wait -1" ::: "memory");
}

// Pack W^T (padded, fp32, contiguous per d) and V (padded, fp16) into workspace.
__global__ __launch_bounds__(256)
void nade_prep(const float* __restrict__ W, const float* __restrict__ V,
               float* __restrict__ Wt, _Float16* __restrict__ Vh) {
  int idx = blockIdx.x * 256 + threadIdx.x;
  if (idx >= NADE_D * NADE_HP) return;
  int d = idx >> 9;
  int k = idx & (NADE_HP - 1);
  float wv = (k < NADE_H) ? W[k * NADE_D + d] : 0.0f;   // W is [H,D]
  float vv = (k < NADE_H) ? V[d * NADE_H + k] : 0.0f;   // V is [D,H]
  Wt[idx] = wv;
  Vh[idx] = (_Float16)vv;
}

__device__ __forceinline__ void nade_step(
    int d, int buf, float xv, float uv,
    const _Float16* __restrict__ Vh, const float* __restrict__ Wt,
    const float* __restrict__ b,
    v2f (&st)[NCHUNK][8], float (*red)[TILE_ROWS * NWAVES],
    int wave, int col, int half, int kbase0,
    float& pR, float& sR) {
  const _Float16* vrow = Vh + (size_t)d * NADE_HP + wave * (NCHUNK * 32);
  const float*    wrow = Wt + (size_t)d * NADE_HP + kbase0;

  // Prefetch rows PFDIST steps ahead; (wave,half,chunk) combos cover all lines.
  int dp = d + PFDIST; dp = (dp < NADE_D) ? dp : (NADE_D - 1);
  const _Float16* vpf = Vh + (size_t)dp * NADE_HP + wave * (NCHUNK * 32);
  const float*    wpf = Wt + (size_t)dp * NADE_HP + kbase0;
  #pragma unroll
  for (int c = 0; c < NCHUNK; ++c) {
    __builtin_prefetch(vpf + c * 32 + half * 8, 0, 3);
    __builtin_prefetch(wpf + c * 32, 0, 3);
  }

  // This step's v_d fragments and w_d values (depend only on d).
  v8h va0[NCHUNK], va1[NCHUNK];
  v4f wq[NCHUNK][4];
  #pragma unroll
  for (int c = 0; c < NCHUNK; ++c) {
    // A operand, contiguous k order used here (same as B): half 0 -> K {0..15}, half 1 -> K {16..31}
    va0[c] = *(const v8h*)(vrow + c * 32 + half * 16);
    va1[c] = *(const v8h*)(vrow + c * 32 + half * 16 + 8);
    #pragma unroll
    for (int q = 0; q < 4; ++q)
      wq[c][q] = *(const v4f*)(wrow + c * 32 + 4 * q);
  }
  float bd = b[d];

  // Build ALL hb fragments first (relu in f16: one v_cvt_pk_rtz + one
  // v_pk_max_num_f16 per pair; cvt results are canonical so no quieting).
  const v2hf z2 = {(__fp16)0.0f, (__fp16)0.0f};
  v16h hbv[NCHUNK];
  #pragma unroll
  for (int c = 0; c < NCHUNK; ++c)
    #pragma unroll
    for (int p = 0; p < 8; ++p) {
      v2hf pk = __builtin_amdgcn_cvt_pkrtz(st[c][p].x, st[c][p].y);
      v2hf r  = __builtin_elementwise_max(pk, z2);
      v2h  rr = __builtin_bit_cast(v2h, r);
      hbv[c][2 * p] = rr.x; hbv[c][2 * p + 1] = rr.y;
    }

  // Then the 4 independent WMMAs back-to-back (no D->A/B hazards between
  // them), then one tree sum. A = v_d broadcast across M, so every lane's
  // acc[0] == dot(relu(a_row), v_d) for batch row `col`.
  v8f accs[NCHUNK];
  #pragma unroll
  for (int c = 0; c < NCHUNK; ++c) {
    v16h va;
    #pragma unroll
    for (int i = 0; i < 8; ++i) { va[i] = va0[c][i]; va[8 + i] = va1[c][i]; }
    const v8f zc = {0.f, 0.f, 0.f, 0.f, 0.f, 0.f, 0.f, 0.f};
    accs[c] = __builtin_amdgcn_wmma_f32_16x16x32_f16(false, va, false, hbv[c],
                                                     (short)0, zc, false, false);
  }
  float part = (accs[0][0] + accs[1][0]) + (accs[2][0] + accs[3][0]);

  // Cross-wave reduction. No lane guard: lanes L and L+16 hold identical
  // values and write the same address (benign). One b128 LDS read per lane.
  red[buf][col * NWAVES + wave] = part;
  wg_barrier_ds_only();
  float4 r4 = *(const float4*)&red[buf][col * NWAVES];
  float t  = (r4.x + r4.y) + (r4.z + r4.w) + bd;
  float p  = __builtin_amdgcn_rcpf(1.0f + __expf(-t));
  float s  = (uv < p) ? 1.0f : 0.0f;
  float xd = (xv == -1.0f) ? s : xv;
  pR = p; sR = s;

  // Rank-1 update: a += xd * W[:,d] as packed FMA (v_pk_fma_f32).
  v2f xd2; xd2.x = xd; xd2.y = xd;
  #pragma unroll
  for (int c = 0; c < NCHUNK; ++c)
    #pragma unroll
    for (int q = 0; q < 4; ++q) {
      v2f lo = __builtin_shufflevector(wq[c][q], wq[c][q], 0, 1);
      v2f hi = __builtin_shufflevector(wq[c][q], wq[c][q], 2, 3);
      st[c][2 * q + 0] = __builtin_elementwise_fma(xd2, lo, st[c][2 * q + 0]);
      st[c][2 * q + 1] = __builtin_elementwise_fma(xd2, hi, st[c][2 * q + 1]);
    }
}

__global__ __launch_bounds__(THREADS)
void nade_scan(const float* __restrict__ x, const float* __restrict__ u,
               const float* __restrict__ cvec, const float* __restrict__ b,
               const float* __restrict__ Wt, const _Float16* __restrict__ Vh,
               float* __restrict__ outP, float* __restrict__ outS) {
  __shared__ float red[2][TILE_ROWS * NWAVES];

  const int lane = threadIdx.x & 31;
  const int wave = threadIdx.x >> 5;
  const int col  = lane & 15;   // batch row within tile (B-fragment column)
  const int half = lane >> 4;   // K-half within a 32-K chunk
  const int row  = blockIdx.x * TILE_ROWS + col;

  // State a[row][h] in WMMA B-fragment layout, stored as float2 pairs:
  // lane holds h = wave*128 + c*32 + half*16 + 2*j (+1).
  v2f st[NCHUNK][8];
  const int kbase0 = wave * (NCHUNK * 32) + half * 16;
  #pragma unroll
  for (int c = 0; c < NCHUNK; ++c)
    #pragma unroll
    for (int j = 0; j < 8; ++j) {
      int h = kbase0 + c * 32 + 2 * j;
      v2f v; v.x = (h < NADE_H) ? cvec[h] : 0.0f;
      v.y = (h + 1 < NADE_H) ? cvec[h + 1] : 0.0f;
      st[c][j] = v;
    }

  const float* xrow = x + (size_t)row * NADE_D;
  const float* urow = u + (size_t)row * NADE_D;
  float* prow = outP + (size_t)row * NADE_D;
  float* srow = outS + (size_t)row * NADE_D;

  for (int d = 0; d < NADE_D; d += 2) {
    float2 xv2 = *(const float2*)(xrow + d);
    float2 uv2 = *(const float2*)(urow + d);
    float p0, s0, p1, s1;
    nade_step(d,     0, xv2.x, uv2.x, Vh, Wt, b, st, red,
              wave, col, half, kbase0, p0, s0);
    nade_step(d + 1, 1, xv2.y, uv2.y, Vh, Wt, b, st, red,
              wave, col, half, kbase0, p1, s1);
    if (wave == 0) {   // uniform branch; duplicate lanes store identical data
      float2 pv; pv.x = p0; pv.y = p1;
      float2 sv; sv.x = s0; sv.y = s1;
      *(float2*)(prow + d) = pv;
      *(float2*)(srow + d) = sv;
    }
  }
}

extern "C" void kernel_launch(void* const* d_in, const int* in_sizes, int n_in,
                              void* d_out, int out_size, void* d_ws, size_t ws_size,
                              hipStream_t stream) {
  const float* x = (const float*)d_in[0];
  const float* u = (const float*)d_in[1];
  const float* W = (const float*)d_in[2];
  const float* c = (const float*)d_in[3];
  const float* V = (const float*)d_in[4];
  const float* b = (const float*)d_in[5];
  float* out = (float*)d_out;

  float*    Wt = (float*)d_ws;                                         // 784*512 f32
  _Float16* Vh = (_Float16*)((char*)d_ws +
                             (size_t)NADE_D * NADE_HP * sizeof(float)); // 784*512 f16

  const int prepN = NADE_D * NADE_HP;
  nade_prep<<<(prepN + 255) / 256, 256, 0, stream>>>(W, V, Wt, Vh);
  nade_scan<<<NADE_B / TILE_ROWS, THREADS, 0, stream>>>(
      x, u, c, b, Wt, Vh, out, out + (size_t)NADE_B * NADE_D);
}
